// AdditiveCoupling_31052613550479
// MI455X (gfx1250) — matrix operand order repaired and checked by reading, not compiled, not run on hardware
//
#include <hip/hip_runtime.h>
#include <hip/hip_bf16.h>
#include <stdint.h>

typedef _Float16 v8h  __attribute__((ext_vector_type(8)));
typedef _Float16 v16h __attribute__((ext_vector_type(16)));
typedef float    v8f  __attribute__((ext_vector_type(8)));

#define B_ROWS   4096
#define D_FULL   784
#define D_HALF   392       // real first/second width
#define KP_IN    448       // 392 padded to mult of 64 (BK)
#define MID      1000
#define MIDP     1024      // 1000 padded
#define NOUTP    512       // 392 padded to mult of 128 (final GEMM N tiles)

// ---------------------------------------------------------------------------
// Pack kernels: f32 -> f16, zero padding, weight transpose to N-major
// ---------------------------------------------------------------------------

__global__ void pack_first(const float* __restrict__ x, float* __restrict__ y,
                           _Float16* __restrict__ A0) {
    int idx = blockIdx.x * 256 + threadIdx.x;
    if (idx >= B_ROWS * KP_IN) return;
    int m = idx / KP_IN, k = idx % KP_IN;
    float v = 0.f;
    if (k < D_HALF) {
        v = x[m * D_FULL + 2 * k];
        y[m * D_FULL + 2 * k] = v;           // first half passes through
    }
    A0[idx] = (_Float16)v;
}

// W_in (392,1000) row-major -> Wt_in[n*448+k], n<1024
__global__ void pack_win(const float* __restrict__ W, _Float16* __restrict__ Wt) {
    int idx = blockIdx.x * 256 + threadIdx.x;
    if (idx >= MIDP * KP_IN) return;
    int n = idx / KP_IN, k = idx % KP_IN;
    Wt[idx] = (n < MID && k < D_HALF) ? (_Float16)W[k * MID + n] : (_Float16)0.f;
}

// W_hid (4,1000,1000) -> Wt_hid[l][n*1024+k]
__global__ void pack_whid(const float* __restrict__ W, _Float16* __restrict__ Wt) {
    int idx = blockIdx.x * 256 + threadIdx.x;
    if (idx >= 4 * MIDP * MIDP) return;
    int l = idx >> 20;
    int n = (idx >> 10) & 1023;
    int k = idx & 1023;
    Wt[idx] = (n < MID && k < MID) ? (_Float16)W[l * MID * MID + k * MID + n]
                                   : (_Float16)0.f;
}

// W_out (1000,392) -> Wt_out[n*1024+k], n<512
__global__ void pack_wout(const float* __restrict__ W, _Float16* __restrict__ Wt) {
    int idx = blockIdx.x * 256 + threadIdx.x;
    if (idx >= NOUTP * MIDP) return;
    int n = idx >> 10, k = idx & 1023;
    Wt[idx] = (n < D_HALF && k < MID) ? (_Float16)W[k * D_HALF + n] : (_Float16)0.f;
}

// bp[0..1023]=b_in padded, bp[1024*(1+l)..]=b_hid[l] padded
__global__ void pack_bias(const float* __restrict__ b_in, const float* __restrict__ b_hid,
                          float* __restrict__ bp) {
    int idx = blockIdx.x * 256 + threadIdx.x;
    if (idx >= 5 * MIDP) return;
    int l = idx >> 10, j = idx & 1023;
    float v = 0.f;
    if (j < MID) v = (l == 0) ? b_in[j] : b_hid[(l - 1) * MID + j];
    bp[idx] = v;
}

__global__ void copy_logdet(const float* __restrict__ src, float* __restrict__ y) {
    y[B_ROWS * D_FULL] = src[0];
}

// ---------------------------------------------------------------------------
// WMMA mainloop: BK=64, async global->LDS, double-buffered, branchless prefetch
// ---------------------------------------------------------------------------
union Frag16 { v16h v; v8h h[2]; };

#define BK         64
#define LDS_STRIDE 72                              // 64 + 8 halves; 144 B rows
#define LDS_BUF_BYTES (128 * LDS_STRIDE * 2)       // 18432 B per buffer

__device__ __forceinline__ uint32_t lds_off(const void* p) {
    // generic LDS pointer: addr[31:0] is the wave-relative LDS byte offset
    return (uint32_t)(uintptr_t)p;
}

__device__ __forceinline__ void async_b128(uint32_t lds, uint64_t gaddr) {
    asm volatile("global_load_async_to_lds_b128 %0, %1, off"
                 :: "v"(lds), "v"(gaddr) : "memory");
}

__device__ __forceinline__ void wmma_mainloop(
    const _Float16* __restrict__ A, const _Float16* __restrict__ Wt,
    int Kpad, int bm, int bn,
    _Float16 (&As)[2][128][LDS_STRIDE], _Float16 (&Bs)[2][128][LDS_STRIDE],
    v8f (&acc)[4][2])
{
    const int tid  = threadIdx.x;
    const int lane = tid & 31;
    const int w    = tid >> 5;
    const int wm   = w >> 2;                 // 0..1  -> M offset wm*64
    const int wn   = w & 3;                  // 0..3  -> N offset wn*32
    const int r    = lane & 15;
    const int kh   = lane >> 4;

    // staging: 128x64 tile = 1024 chunks of 8 halves; 4 per thread per matrix
    uint64_t gA[4], gB[4];
    uint32_t lA[4], lB[4];
#pragma unroll
    for (int i = 0; i < 4; i++) {
        int c   = tid + i * 256;             // 0..1023
        int row = c >> 3;
        int col = (c & 7) * 8;
        gA[i] = (uint64_t)(uintptr_t)(A  + (bm + row) * Kpad + col);
        gB[i] = (uint64_t)(uintptr_t)(Wt + (bn + row) * Kpad + col);
        lA[i] = lds_off(&As[0][row][col]);
        lB[i] = lds_off(&Bs[0][row][col]);
    }

    const int T = Kpad >> 6;

    // prologue: tile 0 -> buffer 0  (ASYNCcnt = 8)
#pragma unroll
    for (int i = 0; i < 4; i++) async_b128(lA[i], gA[i]);
#pragma unroll
    for (int i = 0; i < 4; i++) async_b128(lB[i], gB[i]);

    for (int t = 0; t < T; ++t) {
        const int buf = t & 1;
        // branchless prefetch: last iteration redundantly re-fetches tile T-1
        // into the dead buffer (no readers; previous readers drained at the
        // trailing barrier of t-1; s_endpgm wait-idle drains the leftovers)
        const int      tp = (t + 1 < T) ? t + 1 : T - 1;
        const uint32_t bo = ((t + 1) & 1) ? LDS_BUF_BYTES : 0;
        const uint64_t go = (uint64_t)tp * (BK * 2);
#pragma unroll
        for (int i = 0; i < 4; i++) async_b128(lA[i] + bo, gA[i] + go);
#pragma unroll
        for (int i = 0; i < 4; i++) async_b128(lB[i] + bo, gB[i] + go);
        // tile t's 8 asyncs complete (in-order); prefetch's 8 stay in flight
        asm volatile("s_wait_asynccnt 0x8" ::: "memory");
        __syncthreads();   // all waves' tile-t data visible in LDS

#pragma unroll
        for (int kk = 0; kk < BK; kk += 32) {
            Frag16 af[4], bf[2];
#pragma unroll
            for (int mt = 0; mt < 4; mt++) {
                int row = wm * 64 + mt * 16 + r;
                af[mt].h[0] = *(const v8h*)&As[buf][row][kk + kh * 8];
                af[mt].h[1] = *(const v8h*)&As[buf][row][kk + 16 + kh * 8];
            }
#pragma unroll
            for (int nt = 0; nt < 2; nt++) {
                int row = wn * 32 + nt * 16 + r;
                bf[nt].h[0] = *(const v8h*)&Bs[buf][row][kk + kh * 8];
                bf[nt].h[1] = *(const v8h*)&Bs[buf][row][kk + 16 + kh * 8];
            }
#pragma unroll
            for (int mt = 0; mt < 4; mt++)
#pragma unroll
                for (int nt = 0; nt < 2; nt++)
                    acc[mt][nt] = __builtin_amdgcn_wmma_f32_16x16x32_f16(
                        false, af[mt].v, false, bf[nt].v,
                        (short)0, acc[mt][nt], false, false);
        }
        __syncthreads();   // reads of buf done before it is refilled at t+2
    }
}

// ---------------------------------------------------------------------------
// C(f16) = relu(A(f16, MxK) * Wt(f16, N-major) + bias); tiles 128x128, BK=64
// ---------------------------------------------------------------------------
__global__ __launch_bounds__(256)
void gemm_relu_f16(const _Float16* __restrict__ A, const _Float16* __restrict__ Wt,
                   const float* __restrict__ bias, _Float16* __restrict__ C,
                   int Kpad, int ldc) {
    __shared__ _Float16 As[2][128][LDS_STRIDE];
    __shared__ _Float16 Bs[2][128][LDS_STRIDE];

    const int tid  = threadIdx.x;
    const int lane = tid & 31;
    const int w    = tid >> 5;
    const int wm   = w >> 2, wn = w & 3;
    const int bm   = blockIdx.x * 128, bn = blockIdx.y * 128;
    const int r    = lane & 15, kh = lane >> 4;

    v8f acc[4][2] = {};
    wmma_mainloop(A, Wt, Kpad, bm, bn, As, Bs, acc);

    // Epilogue: bias + ReLU; lane holds col r, rows kh*8+i (C layout table)
#pragma unroll
    for (int mt = 0; mt < 4; mt++)
#pragma unroll
        for (int nt = 0; nt < 2; nt++) {
            int n = bn + wn * 32 + nt * 16 + r;
            float bv = bias[n];
#pragma unroll
            for (int i = 0; i < 8; i++) {
                int m = bm + wm * 64 + mt * 16 + kh * 8 + i;
                float v = acc[mt][nt][i] + bv;
                v = v > 0.f ? v : 0.f;
                C[m * ldc + n] = (_Float16)v;
            }
        }
}

// Final layer: shift = A*Wt_out + b_out; y[m,2n+1] = x[m,2n+1] + shift[m,n]
__global__ __launch_bounds__(256)
void gemm_final(const _Float16* __restrict__ A, const _Float16* __restrict__ Wt,
                const float* __restrict__ b_out, const float* __restrict__ x,
                float* __restrict__ y, int Kpad) {
    __shared__ _Float16 As[2][128][LDS_STRIDE];
    __shared__ _Float16 Bs[2][128][LDS_STRIDE];

    const int tid  = threadIdx.x;
    const int lane = tid & 31;
    const int w    = tid >> 5;
    const int wm   = w >> 2, wn = w & 3;
    const int bm   = blockIdx.x * 128, bn = blockIdx.y * 128;
    const int r    = lane & 15, kh = lane >> 4;

    v8f acc[4][2] = {};
    wmma_mainloop(A, Wt, Kpad, bm, bn, As, Bs, acc);

#pragma unroll
    for (int mt = 0; mt < 4; mt++)
#pragma unroll
        for (int nt = 0; nt < 2; nt++) {
            int n = bn + wn * 32 + nt * 16 + r;
            if (n < D_HALF) {
                float bv = b_out[n];
#pragma unroll
                for (int i = 0; i < 8; i++) {
                    int m = bm + wm * 64 + mt * 16 + kh * 8 + i;
                    float shift = acc[mt][nt][i] + bv;
                    y[m * D_FULL + 2 * n + 1] = x[m * D_FULL + 2 * n + 1] + shift;
                }
            }
        }
}

// ---------------------------------------------------------------------------

extern "C" void kernel_launch(void* const* d_in, const int* in_sizes, int n_in,
                              void* d_out, int out_size, void* d_ws, size_t ws_size,
                              hipStream_t stream) {
    const float* x       = (const float*)d_in[0];
    const float* log_det = (const float*)d_in[1];
    const float* W_in    = (const float*)d_in[2];
    const float* b_in    = (const float*)d_in[3];
    const float* W_hid   = (const float*)d_in[4];
    const float* b_hid   = (const float*)d_in[5];
    const float* W_out   = (const float*)d_in[6];
    const float* b_out   = (const float*)d_in[7];
    float* y = (float*)d_out;

    // Workspace carve-up (all chunk sizes multiples of 256B)
    char* p = (char*)d_ws;
    _Float16* A0     = (_Float16*)p; p += (size_t)B_ROWS * KP_IN * 2;   // 3.5 MB
    _Float16* H1     = (_Float16*)p; p += (size_t)B_ROWS * MIDP * 2;    // 8 MB
    _Float16* H2     = (_Float16*)p; p += (size_t)B_ROWS * MIDP * 2;    // 8 MB
    _Float16* Wt_in  = (_Float16*)p; p += (size_t)MIDP * KP_IN * 2;     // 0.88 MB
    _Float16* Wt_hid = (_Float16*)p; p += (size_t)4 * MIDP * MIDP * 2;  // 8 MB
    _Float16* Wt_out = (_Float16*)p; p += (size_t)NOUTP * MIDP * 2;     // 1 MB
    float*    bp     = (float*)p;    p += (size_t)5 * MIDP * 4;

    // Pack / convert
    pack_first<<<(B_ROWS * KP_IN + 255) / 256, 256, 0, stream>>>(x, y, A0);
    pack_win  <<<(MIDP * KP_IN + 255) / 256, 256, 0, stream>>>(W_in, Wt_in);
    pack_whid <<<(4 * MIDP * MIDP + 255) / 256, 256, 0, stream>>>(W_hid, Wt_hid);
    pack_wout <<<(NOUTP * MIDP + 255) / 256, 256, 0, stream>>>(W_out, Wt_out);
    pack_bias <<<(5 * MIDP + 255) / 256, 256, 0, stream>>>(b_in, b_hid, bp);
    copy_logdet<<<1, 1, 0, stream>>>(log_det, y);

    // MLP chain: 4096 x {448 -> 1024 -> 1024 x4 -> 512(392)}
    dim3 gHid(B_ROWS / 128, MIDP / 128);
    gemm_relu_f16<<<gHid, 256, 0, stream>>>(A0, Wt_in,                  bp,            H1, KP_IN, MIDP);
    gemm_relu_f16<<<gHid, 256, 0, stream>>>(H1, Wt_hid + 0 * MIDP*MIDP, bp + 1*MIDP,  H2, MIDP, MIDP);
    gemm_relu_f16<<<gHid, 256, 0, stream>>>(H2, Wt_hid + 1 * MIDP*MIDP, bp + 2*MIDP,  H1, MIDP, MIDP);
    gemm_relu_f16<<<gHid, 256, 0, stream>>>(H1, Wt_hid + 2 * MIDP*MIDP, bp + 3*MIDP,  H2, MIDP, MIDP);
    gemm_relu_f16<<<gHid, 256, 0, stream>>>(H2, Wt_hid + 3 * MIDP*MIDP, bp + 4*MIDP,  H1, MIDP, MIDP);

    dim3 gOut(B_ROWS / 128, NOUTP / 128);
    gemm_final<<<gOut, 256, 0, stream>>>(H1, Wt_out, b_out, x, y, MIDP);

    (void)in_sizes; (void)n_in; (void)out_size; (void)ws_size;
}
